// BiTreeLSTM_Backward_60421599920145
// MI455X (gfx1250) — matrix operand order repaired and checked by reading, not compiled, not run on hardware
//
#include <hip/hip_runtime.h>

// ---------------------------------------------------------------------------
// BiTreeLSTM forward on gfx1250 (wave32, WMMA bf16 -> f32 accumulate)
//   FEAT = H = 512, DEPTH = 17, N_NODES = 131071
//   All GEMM operands pre-converted to bf16 in ws (no VALU in the k-loop):
//     - weights (iofux/iofuh/px)           : once per call
//     - features                           : once per call
//     - h_final (bf16 shadow of d_out row) : written by each level's epilogue
//   prev_c stays fp32 in ws (it accumulates across 17 levels).
// ---------------------------------------------------------------------------

#define FEAT   512
#define NLEV   17
#define NNODES ((1 << NLEV) - 1)

// ws layout, ushort (bf16) element offsets
#define WX_U   0u            // iofux_w : 2560*512  = 1310720
#define WH_U   1310720u      // iofuh_w : 1310720
#define PW_U   2621440u      // px_w    : 262144
#define FB_U   2883584u      // feats   : 131071*512 = 67108352
#define HB_U   69991936u     // h bf16  : 65536*512  = 33554432
#define CB_OFF 207092736u    // byte offset of float cbuf[65536*512]

typedef __attribute__((ext_vector_type(16))) __bf16 v16bf;
typedef __attribute__((ext_vector_type(8)))  __bf16 v8bf;
typedef __attribute__((ext_vector_type(8)))  float  v8f;

union ABV { v16bf v; v8bf h[2]; };

__device__ __forceinline__ unsigned short f2bf(float f) {
  unsigned int x = __float_as_uint(f);
  x += 0x7fffu + ((x >> 16) & 1u);          // round to nearest even
  return (unsigned short)(x >> 16);
}

// One lane's A operand slice (16x32 bf16 tile) from a bf16 row-major row.
// Layout: lanes 0-15 hold K {kk+0..15}; lanes 16-31 hold
// K {kk+16..31} (matches B). Two 16-byte loads, no conversion.
__device__ __forceinline__ v16bf load_a_bf(const unsigned short* __restrict__ row,
                                           int kk, int hi) {
  ABV a;
  const int b0 = kk + (hi << 4);
  a.h[0] = *(const v8bf*)(row + b0);
  a.h[1] = *(const v8bf*)(row + b0 + 8);
  return a.v;
}

__device__ __forceinline__ float sigmf(float x) {
  return 1.0f / (1.0f + __expf(-x));
}

// ---------------------------------------------------------------------------
// Prologue 1: fp32 weights -> bf16 in ws
// ---------------------------------------------------------------------------
__global__ __launch_bounds__(256) void convert_weights(
    const float* __restrict__ wx, const float* __restrict__ wh,
    const float* __restrict__ pw, unsigned short* __restrict__ dst) {
  const int i = blockIdx.x * 256 + threadIdx.x;
  if (i < 2560 * 512) {
    dst[WX_U + i] = f2bf(wx[i]);
    dst[WH_U + i] = f2bf(wh[i]);
  }
  if (i < 512 * 512) dst[PW_U + i] = f2bf(pw[i]);
}

// ---------------------------------------------------------------------------
// Prologue 2: fp32 features -> bf16 in ws (4 elements / thread, b128 in,
// b64 out)
// ---------------------------------------------------------------------------
__global__ __launch_bounds__(256) void convert_feats(
    const float* __restrict__ f, unsigned short* __restrict__ dst) {
  const size_t base = ((size_t)blockIdx.x * 256 + threadIdx.x) * 4;
  if (base < (size_t)NNODES * FEAT) {
    const float4 v = *(const float4*)(f + base);
    const unsigned int lo = (unsigned int)f2bf(v.x) | ((unsigned int)f2bf(v.y) << 16);
    const unsigned int hi = (unsigned int)f2bf(v.z) | ((unsigned int)f2bf(v.w) << 16);
    *(uint2*)(dst + FB_U + base) = make_uint2(lo, hi);
  }
}

// ---------------------------------------------------------------------------
// One tree level. Block = 128 threads (4 wave32); each wave owns 32 rows x
// 16 j-columns and all six fused outputs (i,o,f,u,r gates + px).
// grid = (ceil(n/128), 32)
// ---------------------------------------------------------------------------
__global__ __launch_bounds__(128) void tree_lstm_level(
    const float* __restrict__ bx, const float* __restrict__ bh,
    const float* __restrict__ pb,
    const unsigned short* __restrict__ wbf,   // ws base (bf16 region)
    float* __restrict__ out, float* __restrict__ cbuf,
    int start, int n, int has_prev, int store_state) {
  const int wave = threadIdx.x >> 5;
  const int lane = threadIdx.x & 31;
  const int hi   = lane >> 4;        // half-wave select
  const int lrow = lane & 15;        // row (A) / column (B,C,D) index
  const int j0   = blockIdx.y << 4;  // 16 j-columns per block
  const int m0   = (blockIdx.x << 7) + (wave << 5);  // 32 rows per wave

  const v8f zero = {0.f, 0.f, 0.f, 0.f, 0.f, 0.f, 0.f, 0.f};
  v8f acc[2][6];                     // [row-tile][i,o,f,u,r,px]
#pragma unroll
  for (int t = 0; t < 2; ++t)
#pragma unroll
    for (int g = 0; g < 6; ++g) acc[t][g] = zero;

  // ---- GEMM 1: feats(bf16) @ [Wx rows g*512+j ; Pw row j]^T, K = 512 ----
  const unsigned short* a0p = wbf + FB_U + ((size_t)(start + m0 + lrow) << 9);
  const unsigned short* a1p = a0p + ((size_t)16 << 9);
  const unsigned short* wrow[6];
#pragma unroll
  for (int g = 0; g < 5; ++g)
    wrow[g] = wbf + WX_U + ((size_t)((g << 9) + j0 + lrow) << 9);
  wrow[5] = wbf + PW_U + ((size_t)(j0 + lrow) << 9);

  for (int kk = 0; kk < FEAT; kk += 32) {
    v16bf a0 = load_a_bf(a0p, kk, hi);
    v16bf a1 = load_a_bf(a1p, kk, hi);
#pragma unroll
    for (int g = 0; g < 6; ++g) {
      // B 32x16: lane = column, 16 contiguous K per half-wave
      v16bf b = *(const v16bf*)(wrow[g] + kk + (hi << 4));
      acc[0][g] = __builtin_amdgcn_wmma_f32_16x16x32_bf16(
          false, a0, false, b, (short)0, acc[0][g], false, false);
      acc[1][g] = __builtin_amdgcn_wmma_f32_16x16x32_bf16(
          false, a1, false, b, (short)0, acc[1][g], false, false);
    }
  }

  // ---- GEMM 2: repeat(prev_h,2)(bf16 shadow) @ Wh^T, K = 512 ----
  if (has_prev) {
    const int p0 = (start + m0 + lrow - 1) >> 1;        // parent node of row
    const int p1 = (start + m0 + 16 + lrow - 1) >> 1;
    const unsigned short* h0p = wbf + HB_U + ((size_t)p0 << 9);
    const unsigned short* h1p = wbf + HB_U + ((size_t)p1 << 9);
    const unsigned short* whrow[5];
#pragma unroll
    for (int g = 0; g < 5; ++g)
      whrow[g] = wbf + WH_U + ((size_t)((g << 9) + j0 + lrow) << 9);

    for (int kk = 0; kk < FEAT; kk += 32) {
      v16bf a0 = load_a_bf(h0p, kk, hi);
      v16bf a1 = load_a_bf(h1p, kk, hi);
#pragma unroll
      for (int g = 0; g < 5; ++g) {
        v16bf b = *(const v16bf*)(whrow[g] + kk + (hi << 4));
        acc[0][g] = __builtin_amdgcn_wmma_f32_16x16x32_bf16(
            false, a0, false, b, (short)0, acc[0][g], false, false);
        acc[1][g] = __builtin_amdgcn_wmma_f32_16x16x32_bf16(
            false, a1, false, b, (short)0, acc[1][g], false, false);
      }
    }
  }

  // ---- epilogue: biases + gates + c/h updates, all in-register ----
  const int j = j0 + lrow;                   // D column for this lane
  float bias[5];
#pragma unroll
  for (int g = 0; g < 5; ++g) bias[g] = bx[(g << 9) + j] + bh[(g << 9) + j];
  const float pbj = pb[j];
  unsigned short* hbf = (unsigned short*)wbf + HB_U;   // bf16 h shadow

#pragma unroll
  for (int t = 0; t < 2; ++t) {
#pragma unroll
    for (int v = 0; v < 8; ++v) {
      const int rl = m0 + (t << 4) + v + (hi << 3);     // D row (local)
      if (rl < n) {
        const int node = start + rl;
        const float iv  = sigmf(acc[t][0][v] + bias[0]);
        const float ov  = sigmf(acc[t][1][v] + bias[1]);
        const float fv  = sigmf(acc[t][2][v] + bias[2]);
        const float uv  = tanhf(acc[t][3][v] + bias[3]);
        const float rv  = sigmf(acc[t][4][v] + bias[4]);
        const float pxv = acc[t][5][v] + pbj;
        const float pc  = has_prev
            ? cbuf[((size_t)((node - 1) >> 1) << 9) + j] : 0.0f;
        const float c  = iv * uv + fv * pc;
        const float h  = ov * tanhf(c);
        const float hf = rv * h + (1.0f - rv) * pxv;
        if (store_state) {
          cbuf[((size_t)node << 9) + j] = c;
          hbf[((size_t)node << 9) + j] = f2bf(hf);
        }
        out[((size_t)node << 9) + j] = hf;
      }
    }
  }
}

// ---------------------------------------------------------------------------
extern "C" void kernel_launch(void* const* d_in, const int* in_sizes, int n_in,
                              void* d_out, int out_size, void* d_ws,
                              size_t ws_size, hipStream_t stream) {
  const float* feats   = (const float*)d_in[0];
  const float* px_w    = (const float*)d_in[1];
  const float* px_b    = (const float*)d_in[2];
  const float* iofux_w = (const float*)d_in[3];
  const float* iofux_b = (const float*)d_in[4];
  const float* iofuh_w = (const float*)d_in[5];
  const float* iofuh_b = (const float*)d_in[6];

  float* out = (float*)d_out;
  unsigned short* wbf = (unsigned short*)d_ws;
  float* cbuf = (float*)((char*)d_ws + CB_OFF);

  convert_weights<<<dim3((2560 * 512 + 255) / 256), dim3(256), 0, stream>>>(
      iofux_w, iofuh_w, px_w, wbf);
  convert_feats<<<dim3(((NNODES * FEAT / 4) + 255) / 256), dim3(256), 0,
                  stream>>>(feats, wbf);

  for (int d = 0; d < NLEV; ++d) {
    const int start = (1 << d) - 1;
    const int n = 1 << d;
    dim3 grid((n + 127) / 128, FEAT / 16);
    tree_lstm_level<<<grid, dim3(128), 0, stream>>>(
        iofux_b, iofuh_b, px_b, wbf, out, cbuf,
        start, n, d > 0 ? 1 : 0, d < NLEV - 1 ? 1 : 0);
  }
}
